// ClusterMemory_30820685316319
// MI455X (gfx1250) — compile-verified, matrix operand order checked by reading, not run on hardware
//
#include <hip/hip_runtime.h>
#include <hip/hip_bf16.h>
#include <math.h>

// Problem constants (from reference)
#define BATCH    1024
#define NFEAT    256
#define NSAMP    100000
#define INV_TEMP 20.0f            // 1 / 0.05

// Tiling
#define FTILES   (NSAMP / 16)     // 6250 feature tiles of 16 rows
#define NSLICE   125              // grid.y slices over feature tiles
#define TPS      (FTILES / NSLICE)// 50 tiles per slice
#define WAVES    8                // waves per block (256 threads, wave32)
#define ROWS_PER_BLOCK (WAVES * 16)  // 128 batch rows per block
#define NPART    NSLICE           // partials per batch row (one per slice)
#define LDSROW   260              // padded f32 LDS row stride (floats)
#define LDSROW16 264              // padded f16 LDS row stride (halves)

typedef __attribute__((ext_vector_type(16))) _Float16 v16h;
typedef __attribute__((ext_vector_type(8)))  _Float16 v8h;
typedef __attribute__((ext_vector_type(4)))  _Float16 v4h;
typedef __attribute__((ext_vector_type(8)))  float    v8f;
typedef __attribute__((ext_vector_type(4)))  int      v4i;

#if defined(__gfx1250__) && __has_builtin(__builtin_amdgcn_global_load_async_to_lds_b128)
#define USE_ASYNC_LDS 1
#else
#define USE_ASYNC_LDS 0
#endif

__device__ __forceinline__ void wait_async_zero() {
#if USE_ASYNC_LDS
#if __has_builtin(__builtin_amdgcn_s_wait_asynccnt)
    __builtin_amdgcn_s_wait_asynccnt(0);
#else
    asm volatile("s_wait_asynccnt 0x0" ::: "memory");
#endif
#endif
}

// Copy one 16-byte chunk global -> LDS (gfx1250 async-to-LDS engine if available).
__device__ __forceinline__ void copy16(float* l, const float* g) {
#if USE_ASYNC_LDS
    typedef __attribute__((address_space(1))) v4i g_v4i;
    typedef __attribute__((address_space(3))) v4i l_v4i;
    __builtin_amdgcn_global_load_async_to_lds_b128(
        (g_v4i*)(uintptr_t)g, (l_v4i*)l, 0, 0);
#else
    *(float4*)l = *(const float4*)g;
#endif
}

// Stage one feature tile (16 rows x 256 f32, contiguous 16KB in global) into
// a padded f32 LDS buffer. 256 threads x 4 x 16B.
__device__ __forceinline__ void stage_tile(float* __restrict__ dst,
                                           const float* __restrict__ src) {
    const int r  = threadIdx.x >> 4;          // 0..15 : feature row in tile
    const int cs = (threadIdx.x & 15) * 4;    // 16B segment within row
    const float* g = src + r * NFEAT + cs;
    float*       l = dst + r * LDSROW + cs;
#pragma unroll
    for (int j = 0; j < 4; ++j)
        copy16(l + j * 64, g + j * 64);
}

// Convert a staged f32 tile to f16 in LDS, once per block (shared by 8 waves).
__device__ __forceinline__ void convert_tile(const float* __restrict__ src32,
                                             _Float16* __restrict__ dst16) {
    const int r  = threadIdx.x >> 4;
    const int cs = (threadIdx.x & 15) * 4;
#pragma unroll
    for (int j = 0; j < 4; ++j) {
        const float4 v = *(const float4*)(src32 + r * LDSROW + cs + j * 64);
        v4h h;
        h[0] = (_Float16)v.x; h[1] = (_Float16)v.y;
        h[2] = (_Float16)v.z; h[3] = (_Float16)v.w;
        *(v4h*)(dst16 + r * LDSROW16 + cs + j * 64) = h;
    }
}

// Pack a 16-bit WMMA operand fragment from a row-major f16 row pointer:
// 8 contiguous K at (khalf) and 8 more at (16 + khalf) -> two b128 loads.
__device__ __forceinline__ v16h load_frag_f16(const _Float16* __restrict__ p,
                                              int khalf) {
    v8h lo = *(const v8h*)(p + khalf);
    v8h hi = *(const v8h*)(p + 16 + khalf);
    return __builtin_shufflevector(lo, hi, 0, 1, 2, 3, 4, 5, 6, 7,
                                           8, 9, 10, 11, 12, 13, 14, 15);
}

// Pack a fragment from a row-major f32 row pointer (used once for B/inputs).
__device__ __forceinline__ v16h load_frag_f32(const float* __restrict__ p, int khalf) {
    const float4* q0 = reinterpret_cast<const float4*>(p + khalf);
    const float4* q1 = reinterpret_cast<const float4*>(p + 16 + khalf);
    float4 a0 = q0[0], a1 = q0[1];
    float4 b0 = q1[0], b1 = q1[1];
    v16h r;
    r[0]  = (_Float16)a0.x; r[1]  = (_Float16)a0.y;
    r[2]  = (_Float16)a0.z; r[3]  = (_Float16)a0.w;
    r[4]  = (_Float16)a1.x; r[5]  = (_Float16)a1.y;
    r[6]  = (_Float16)a1.z; r[7]  = (_Float16)a1.w;
    r[8]  = (_Float16)b0.x; r[9]  = (_Float16)b0.y;
    r[10] = (_Float16)b0.z; r[11] = (_Float16)b0.w;
    r[12] = (_Float16)b1.x; r[13] = (_Float16)b1.y;
    r[14] = (_Float16)b1.z; r[15] = (_Float16)b1.w;
    return r;
}

// Kernel 1: streaming WMMA GEMM + online logsumexp partials.
// Grid: (BATCH/128, NSLICE). Block: 256 threads = 8 waves.
// Pipeline per tile: async-stage(t+1) || 8x WMMA on f16 tile(t), then
// wait_asynccnt + barrier, shared f32->f16 convert(t+1), barrier.
__global__ void __launch_bounds__(256)
nce_partial_kernel(const float* __restrict__ inputs,
                   const float* __restrict__ feats,
                   float* __restrict__ m_part,
                   float* __restrict__ s_part) {
    __shared__ float    stg[2][16 * LDSROW];     // async-staged f32 tiles
    __shared__ _Float16 t16[2][16 * LDSROW16];   // converted f16 tiles

    const int lane  = threadIdx.x & 31;
    const int wave  = threadIdx.x >> 5;
    const int col   = lane & 15;             // batch col / A row within tile
    const int khalf = (lane >> 4) * 8;       // K sub-offset for this lane half

    // Resident B fragments: this wave's 16-row batch tile, full K=256.
    const int   brow = blockIdx.x * ROWS_PER_BLOCK + wave * 16 + col;
    const float* bp  = inputs + (size_t)brow * NFEAT;
    v16h bfrag[8];
#pragma unroll
    for (int kf = 0; kf < 8; ++kf)
        bfrag[kf] = load_frag_f32(bp + kf * 32, khalf);

    float m = -INFINITY;
    float s = 0.0f;

    const int tbeg = blockIdx.y * TPS;

    // Prologue: stage + convert first tile.
    stage_tile(&stg[0][0], feats + (size_t)tbeg * 16 * NFEAT);
    wait_async_zero();
    __syncthreads();
    convert_tile(&stg[0][0], &t16[0][0]);
    __syncthreads();

    for (int i = 0; i < TPS; ++i) {
        const int buf = i & 1;

        // Stage next tile into the other f32 buffer (overlaps WMMAs below).
        if (i + 1 < TPS)
            stage_tile(&stg[buf ^ 1][0],
                       feats + (size_t)(tbeg + i + 1) * 16 * NFEAT);

        // 16x16 logits tile: 8 chained WMMAs over K=256, operands from f16 LDS.
        const _Float16* ap = &t16[buf][col * LDSROW16];
        v8f c = {};
#pragma unroll
        for (int kf = 0; kf < 8; ++kf) {
            v16h a = load_frag_f16(ap + kf * 32, khalf);
            c = __builtin_amdgcn_wmma_f32_16x16x32_f16(
                    /*neg_a=*/false, a, /*neg_b=*/false, bfrag[kf],
                    /*c_mod=*/(short)0, c, /*reuse_a=*/false, /*reuse_b=*/false);
        }

        // Online logsumexp over the 8 sample-logits this lane owns.
        float v[8];
        float tmax = -INFINITY;
#pragma unroll
        for (int j = 0; j < 8; ++j) {
            v[j] = c[j] * INV_TEMP;
            tmax = fmaxf(tmax, v[j]);
        }
        float nm = fmaxf(m, tmax);
        s *= __expf(m - nm);
#pragma unroll
        for (int j = 0; j < 8; ++j)
            s += __expf(v[j] - nm);
        m = nm;

        // Async copy done + everyone finished this tile's compute.
        wait_async_zero();
        __syncthreads();

        // Shared conversion of the next tile, then publish it.
        if (i + 1 < TPS)
            convert_tile(&stg[buf ^ 1][0], &t16[buf ^ 1][0]);
        __syncthreads();
    }

    // Merge lane L with lane L+16 (other 8 sample rows, same batch column).
    {
        float om = __shfl_xor(m, 16, 32);
        float os = __shfl_xor(s, 16, 32);
        float nm = fmaxf(m, om);
        s = s * __expf(m - nm) + os * __expf(om - nm);
        m = nm;
    }

    if (lane < 16) {
        const int row = blockIdx.x * ROWS_PER_BLOCK + wave * 16 + lane;
        m_part[(size_t)row * NPART + blockIdx.y] = m;
        s_part[(size_t)row * NPART + blockIdx.y] = s;
    }
}

// Kernel 2: per-row partial merge + exact f32 target logit + mean NLL.
// Grid: BATCH blocks of one wave (32 threads).
__global__ void __launch_bounds__(32)
nce_reduce_kernel(const float* __restrict__ inputs,
                  const float* __restrict__ feats,
                  const int*   __restrict__ targets,
                  const float* __restrict__ m_part,
                  const float* __restrict__ s_part,
                  float* __restrict__ out) {
    const int row  = blockIdx.x;
    const int lane = threadIdx.x;

    // Merge this row's softmax partials.
    float m = -INFINITY, s = 0.0f;
    for (int p = lane; p < NPART; p += 32) {
        float pm = m_part[(size_t)row * NPART + p];
        float ps = s_part[(size_t)row * NPART + p];
        float nm = fmaxf(m, pm);
        s = s * __expf(m - nm) + ps * __expf(pm - nm);
        m = nm;
    }
#pragma unroll
    for (int off = 16; off > 0; off >>= 1) {
        float om = __shfl_xor(m, off, 32);
        float os = __shfl_xor(s, off, 32);
        float nm = fmaxf(m, om);
        s = s * __expf(m - nm) + os * __expf(om - nm);
        m = nm;
    }

    // Exact f32 target logit.
    const int tgt = targets[row];
    const float* x = inputs + (size_t)row * NFEAT;
    const float* f = feats  + (size_t)tgt * NFEAT;
    float d = 0.0f;
#pragma unroll
    for (int k = lane; k < NFEAT; k += 32)
        d = fmaf(x[k], f[k], d);
#pragma unroll
    for (int off = 16; off > 0; off >>= 1)
        d += __shfl_xor(d, off, 32);

    if (lane == 0) {
        float lse = m + __logf(s);
        float nll = lse - d * INV_TEMP;
        atomicAdd(out, nll * (1.0f / (float)BATCH));
    }
}

extern "C" void kernel_launch(void* const* d_in, const int* in_sizes, int n_in,
                              void* d_out, int out_size, void* d_ws, size_t ws_size,
                              hipStream_t stream) {
    const float* inputs  = (const float*)d_in[0];   // [1024, 256] f32
    const int*   targets = (const int*)  d_in[1];   // [1024] int
    const float* feats   = (const float*)d_in[2];   // [100000, 256] f32
    float* out = (float*)d_out;

    float* m_part = (float*)d_ws;                         // [1024, 125]
    float* s_part = m_part + (size_t)BATCH * NPART;       // [1024, 125]

    (void)hipMemsetAsync(d_out, 0, sizeof(float), stream);

    nce_partial_kernel<<<dim3(BATCH / ROWS_PER_BLOCK, NSLICE), 256, 0, stream>>>(
        inputs, feats, m_part, s_part);

    nce_reduce_kernel<<<BATCH, 32, 0, stream>>>(
        inputs, feats, targets, m_part, s_part, out);
}
